// SpatialAttn_76905684402259
// MI455X (gfx1250) — matrix operand order repaired and checked by reading, not compiled, not run on hardware
//
#include <hip/hip_runtime.h>

typedef __attribute__((ext_vector_type(16))) __bf16 v16bf;
typedef __attribute__((ext_vector_type(8)))  float  v8f;
typedef __attribute__((ext_vector_type(4)))  unsigned int u32x4;
typedef __attribute__((ext_vector_type(8)))  unsigned int u32x8;

union Frag16 {
  v16bf v;
  unsigned short u[16];
  uint4 q[2];
};

__device__ __forceinline__ unsigned short f2bf(float f) {
  unsigned int u = __builtin_bit_cast(unsigned int, f);
  u += 0x7fffu + ((u >> 16) & 1u);            // round-to-nearest-even
  return (unsigned short)(u >> 16);
}
__device__ __forceinline__ float bf2f(unsigned short h) {
  unsigned int u = ((unsigned int)h) << 16;
  return __builtin_bit_cast(float, u);
}

__device__ __forceinline__ v8f wmma_bf16(v16bf a, v16bf b, v8f c) {
  // (neg_a, A, neg_b, B, c_mod, C, reuse_a, reuse_b)
  return __builtin_amdgcn_wmma_f32_16x16x32_bf16(false, a, false, b, (short)0, c,
                                                 false, false);
}

// ---- Tensor Data Mover: 2-D tile load Global -> LDS (CDNA5 TDM) ----
// Descriptor per CDNA5 ISA ch.8: group0 (4 SGPRs) + group1 (8 SGPRs), 2-D form
// (VADDR2/VADDR3 = NULL). data_size_log2: 0=1B,1=2B,2=4B,3=8B. Dims/strides in
// data_size units. One issue per wave; EXEC is ignored by TDM.
__device__ __forceinline__ void tdm_load_2d_to_lds(unsigned lds_off, const void* gaddr,
                                                   unsigned data_size_log2,
                                                   unsigned tile_d0, unsigned tile_d1,
                                                   unsigned tensor_d0, unsigned tensor_d1,
                                                   unsigned stride_d0) {
  unsigned long long ga = (unsigned long long)(size_t)gaddr;
  u32x4 g0;
  g0[0] = 1u;                                               // count=1 (valid D#)
  g0[1] = lds_off;                                          // lds_addr (bytes)
  g0[2] = (unsigned)(ga & 0xffffffffu);                     // global_addr[31:0]
  g0[3] = (unsigned)((ga >> 32) & 0x01ffffffu) | (2u << 30);// addr[56:32] | type=2
  u32x8 g1;
  g1[0] = (data_size_log2 & 3u) << 16;                      // wg_mask=0, data_size
  g1[1] = (tensor_d0 & 0xffffu) << 16;                      // tensor_dim0[15:0]
  g1[2] = ((tensor_d0 >> 16) & 0xffffu) | ((tensor_d1 & 0xffffu) << 16);
  g1[3] = ((tensor_d1 >> 16) & 0xffffu) | ((tile_d0 & 0xffffu) << 16);
  g1[4] = tile_d1 & 0xffffu;                                // tile_dim1, tile_dim2=0
  g1[5] = stride_d0;                                        // tensor_dim0_stride lo
  g1[6] = 0u;                                               // stride hi, dim1_stride
  g1[7] = 0u;
  asm volatile("tensor_load_to_lds %0, %1" :: "s"(g0), "s"(g1) : "memory");
}

// A-matrix 16x32 bf16 fragment from row-major [*, ld] bf16 storage at (m0,k0).
template <typename P>
__device__ __forceinline__ v16bf load_a_frag(const P* p, int ld, int m0, int k0) {
  int lane = threadIdx.x & 31;
  const P* base = p + (size_t)(m0 + (lane & 15)) * ld + k0 + ((lane >> 4) << 3);
  Frag16 f;
  f.q[0] = *(const uint4*)(base);        // K chunk 0 (8 bf16)
  f.q[1] = *(const uint4*)(base + 16);   // K chunk +16
  return f.v;
}

// B-matrix 32x16 bf16 fragment where memory is N-major: element(k,n)=p[n*ld+k].
__device__ __forceinline__ v16bf load_b_frag_nmajor(const unsigned short* p, int ld,
                                                    int k0, int n0) {
  int lane = threadIdx.x & 31;
  const unsigned short* base =
      p + (size_t)(n0 + (lane & 15)) * ld + k0 + ((lane >> 4) << 3);
  Frag16 f;
  f.q[0] = *(const uint4*)(base);        // K chunk 0 (8 bf16)
  f.q[1] = *(const uint4*)(base + 16);   // K chunk +16
  return f.v;
}

// B-matrix 32x16 fragment where memory is K-major: element(k,n)=p[k*ld+n].
template <typename P>
__device__ __forceinline__ v16bf load_b_frag_kmajor(const P* p, int ld, int k0, int n0) {
  int lane = threadIdx.x & 31;
  int n = n0 + (lane & 15);
  int kc = k0 + ((lane >> 4) << 3);
  Frag16 f;
#pragma unroll
  for (int i = 0; i < 16; ++i) f.u[i] = p[(kc + i + ((i >> 3) << 3)) * ld + n];
  return f.v;
}

// A-fragment of gd^T: A(m=j,k=i) = gd[i*128 + j], fp32 source -> bf16.
__device__ __forceinline__ v16bf load_a_frag_gdT(const float* g, int m0, int k0) {
  int lane = threadIdx.x & 31;
  int m = m0 + (lane & 15);
  int kc = k0 + ((lane >> 4) << 3);
  Frag16 f;
#pragma unroll
  for (int i = 0; i < 8; ++i) {
    f.u[i]     = f2bf(g[(kc + i) * 128 + m]);
    f.u[8 + i] = f2bf(g[(kc + 16 + i) * 128 + m]);
  }
  return f.v;
}

// ---------------- Stage 0: pack weights to bf16 [384][128] ----------------
__global__ void __launch_bounds__(256) pack_w_kernel(const float* __restrict__ wA,
                                                     const float* __restrict__ wB,
                                                     const float* __restrict__ wV,
                                                     unsigned short* __restrict__ Wall) {
  int idx = blockIdx.x * 256 + threadIdx.x;   // 0..49151
  int m = idx >> 7;
  int k = idx & 127;
  const float* src = (m < 128) ? wA : (m < 256) ? wB : wV;
  Wall[idx] = f2bf(src[(m & 127) * 128 + k]);
}

// ---------- Stage 1: A/B/V = W@x + bias, bf16 outputs (WMMA) ----------
__global__ void __launch_bounds__(256) qkv_kernel(
    const float* __restrict__ x, const unsigned short* __restrict__ Wall,
    const float* __restrict__ bA, const float* __restrict__ bB,
    const float* __restrict__ bV, unsigned short* __restrict__ Abf,
    unsigned short* __restrict__ Bbf, unsigned short* __restrict__ Vbf) {
  __shared__ unsigned short xs[128 * 64];
  const int b = blockIdx.y;
  const int n0blk = blockIdx.x * 64;
  const int tid = threadIdx.x;
  const float* xb = x + ((size_t)b << 19);
  for (int idx = tid; idx < 128 * 64; idx += 256) {
    int c = idx >> 6, n = idx & 63;
    xs[idx] = f2bf(xb[((size_t)c << 12) + n0blk + n]);
  }
  __syncthreads();

  const int wave = tid >> 5, lane = tid & 31;
  const int nt = wave & 3, mg = wave >> 2;
  const int ncol = n0blk + nt * 16 + (lane & 15);
  // Outer loop over the three output matrices (compile-time select after
  // unrolling -> no divergent exec-mask control flow in the hot loop).
#pragma unroll
  for (int mat = 0; mat < 3; ++mat) {
    unsigned short* const dst0 = (mat == 0) ? Abf : (mat == 1) ? Bbf : Vbf;
    const float* const bias    = (mat == 0) ? bA  : (mat == 1) ? bB  : bV;
    unsigned short* const dst  = dst0 + ((size_t)b << 19) + ncol;
    for (int i = 0; i < 4; ++i) {
      const int mtl = mg + 2 * i;            // local M-tile within this matrix
      const int mt  = mat * 8 + mtl;         // global M-tile in Wall
      v8f acc = {};
#pragma unroll
      for (int kt = 0; kt < 4; ++kt) {
        v16bf a  = load_a_frag(Wall, 128, mt * 16, kt * 32);
        v16bf bb = load_b_frag_kmajor(xs, 64, kt * 32, nt * 16);
        acc = wmma_bf16(a, bb, acc);
      }
#pragma unroll
      for (int r = 0; r < 8; ++r) {
        int mrow = mtl * 16 + r + ((lane >> 4) << 3);
        dst[(size_t)mrow << 12] = f2bf(acc[r] + bias[mrow]);
      }
    }
  }
}

// ---------- Stage 2: in-place row softmax over 4096, bf16 ----------
__global__ void __launch_bounds__(256) softmax_kernel(unsigned short* __restrict__ buf,
                                                      float scale) {
  __shared__ float red[8];
  unsigned short* p = buf + ((size_t)blockIdx.x << 12);
  const int tid = threadIdx.x;
  float v[16];
  float m = -3.0e38f;
#pragma unroll
  for (int i = 0; i < 16; ++i) {
    v[i] = bf2f(p[tid + (i << 8)]);
    m = fmaxf(m, v[i]);
  }
#pragma unroll
  for (int off = 16; off; off >>= 1) m = fmaxf(m, __shfl_xor(m, off, 32));
  if ((tid & 31) == 0) red[tid >> 5] = m;
  __syncthreads();
  m = red[0];
#pragma unroll
  for (int w = 1; w < 8; ++w) m = fmaxf(m, red[w]);

  float s = 0.f;
#pragma unroll
  for (int i = 0; i < 16; ++i) {
    v[i] = __expf(v[i] - m);
    s += v[i];
  }
#pragma unroll
  for (int off = 16; off; off >>= 1) s += __shfl_xor(s, off, 32);
  __syncthreads();
  if ((tid & 31) == 0) red[tid >> 5] = s;
  __syncthreads();
  s = 0.f;
#pragma unroll
  for (int w = 0; w < 8; ++w) s += red[w];

  const float inv = scale / s;
#pragma unroll
  for (int i = 0; i < 16; ++i) p[tid + (i << 8)] = f2bf(v[i] * inv);
}

// ---------- Stage 3: gd[b] = smB[b] @ A[b]^T  (128x128, K=4096) ----------
__global__ void __launch_bounds__(256) gd_kernel(const unsigned short* __restrict__ smB,
                                                 const unsigned short* __restrict__ Abf,
                                                 float* __restrict__ gd) {
  const int b = blockIdx.x;
  const int wave = threadIdx.x >> 5, lane = threadIdx.x & 31;
  const unsigned short* sb = smB + ((size_t)b << 19);
  const unsigned short* ab = Abf + ((size_t)b << 19);
  v8f acc[8];
#pragma unroll
  for (int j = 0; j < 8; ++j) { v8f z = {}; acc[j] = z; }
  const int m0 = wave * 16;
  for (int kt = 0; kt < 128; ++kt) {
    if ((kt & 7) == 0 && kt + 8 < 128) {
      // global_prefetch_b8 ahead on the streamed A-operand rows
      __builtin_prefetch(sb + (size_t)(m0 + (threadIdx.x & 15)) * 4096 + (kt + 8) * 32,
                         0, 1);
    }
    v16bf a = load_a_frag(sb, 4096, m0, kt * 32);
#pragma unroll
    for (int ntile = 0; ntile < 8; ++ntile) {
      v16bf bb = load_b_frag_nmajor(ab, 4096, kt * 32, ntile * 16);
      acc[ntile] = wmma_bf16(a, bb, acc[ntile]);
    }
  }
  float* g = gd + ((size_t)b << 14);
#pragma unroll
  for (int ntile = 0; ntile < 8; ++ntile) {
    int n = ntile * 16 + (lane & 15);
#pragma unroll
    for (int r = 0; r < 8; ++r) {
      int m = m0 + r + ((lane >> 4) << 3);
      g[m * 128 + n] = acc[ntile][r];
    }
  }
}

// ---------- Stage 4: out[b] = gd[b]^T @ av[b]  (128x4096, K=128) ----------
__global__ void __launch_bounds__(256) out_kernel(const float* __restrict__ gd,
                                                  const unsigned short* __restrict__ av,
                                                  float* __restrict__ out) {
  __shared__ unsigned short vs[128 * 64];
  const int b = blockIdx.y;
  const int n0blk = blockIdx.x * 64;
  const int tid = threadIdx.x;
  const unsigned short* avb = av + ((size_t)b << 19) + n0blk;

  // Stage the 128x64 bf16 av-tile into LDS with the Tensor Data Mover:
  // tile_dim0=64 (contiguous cols), tile_dim1=128 rows, row stride 4096 elems.
  if (tid < 32) {
    tdm_load_2d_to_lds((unsigned)(size_t)(void*)vs, avb,
                       /*data_size_log2=*/1,
                       /*tile_d0=*/64, /*tile_d1=*/128,
                       /*tensor_d0=*/4096, /*tensor_d1=*/128,
                       /*stride_d0=*/4096);
    __builtin_amdgcn_s_wait_tensorcnt(0);
  }
  __syncthreads();

  const float* g = gd + ((size_t)b << 14);
  const int wave = tid >> 5, lane = tid & 31;
  const int nt = wave & 3, mg = wave >> 2;
  const int ncol = n0blk + nt * 16 + (lane & 15);
  float* ob = out + ((size_t)b << 19);
  for (int i = 0; i < 4; ++i) {
    int mt = mg + 2 * i;                    // 8 M-tiles over 2 wave groups
    v8f acc = {};
#pragma unroll
    for (int kt = 0; kt < 4; ++kt) {
      v16bf a  = load_a_frag_gdT(g, mt * 16, kt * 32);
      v16bf bb = load_b_frag_kmajor(vs, 64, kt * 32, nt * 16);
      acc = wmma_bf16(a, bb, acc);
    }
#pragma unroll
    for (int r = 0; r < 8; ++r) {
      int j = mt * 16 + r + ((lane >> 4) << 3);
      ob[((size_t)j << 12) + ncol] = acc[r];
    }
  }
}

extern "C" void kernel_launch(void* const* d_in, const int* in_sizes, int n_in,
                              void* d_out, int out_size, void* d_ws, size_t ws_size,
                              hipStream_t stream) {
  const float* x  = (const float*)d_in[0];
  const float* wA = (const float*)d_in[1];
  const float* bA = (const float*)d_in[2];
  const float* wB = (const float*)d_in[3];
  const float* bB = (const float*)d_in[4];
  const float* wV = (const float*)d_in[5];
  const float* bV = (const float*)d_in[6];
  float* out = (float*)d_out;

  char* ws = (char*)d_ws;
  const size_t SZ_BCHW_BF = (size_t)32 * 128 * 4096 * 2;  // 32 MB per tensor
  unsigned short* Wall = (unsigned short*)(ws);                              // 96 KB
  unsigned short* Abf  = (unsigned short*)(ws + (1 << 20));
  unsigned short* Bbf  = (unsigned short*)(ws + (1 << 20) + SZ_BCHW_BF);
  unsigned short* Vbf  = (unsigned short*)(ws + (1 << 20) + 2 * SZ_BCHW_BF);
  float*          gd   = (float*)(ws + (1 << 20) + 3 * SZ_BCHW_BF);          // 2 MB

  pack_w_kernel<<<192, 256, 0, stream>>>(wA, wB, wV, Wall);
  qkv_kernel<<<dim3(64, 32), 256, 0, stream>>>(x, Wall, bA, bB, bV, Abf, Bbf, Vbf);
  softmax_kernel<<<4096, 256, 0, stream>>>(Bbf, 1.0f / 4096.0f);  // folds /hw into gd
  softmax_kernel<<<4096, 256, 0, stream>>>(Vbf, 1.0f);
  gd_kernel<<<32, 256, 0, stream>>>(Bbf, Abf, gd);
  out_kernel<<<dim3(64, 32), 256, 0, stream>>>(gd, Vbf, out);
}
